// TripletLoss_10488310136948
// MI455X (gfx1250) — matrix operand order repaired and checked by reading, not compiled, not run on hardware
//
#include <hip/hip_runtime.h>
#include <math.h>

typedef __attribute__((ext_vector_type(16))) _Float16 v16h;
typedef __attribute__((ext_vector_type(2)))  _Float16 h2;
typedef __attribute__((ext_vector_type(8)))  float    v8f;

#define B_      8
#define C_      64
#define N_      2000
#define M_      8
#define W_      240
#define WH_     (240 * 240)
#define PTS     16
#define TILES   (N_ / PTS)          /* 125 */
#define NBLK    (B_ * TILES)        /* 1000 */
#define EPS_    1e-7f
#define MARGIN_ 0.5f

// One wave32 per 16 points. Each lane owns channels (2*lane, 2*lane+1) for all
// 16 points: f1 lives in registers, coordinates broadcast via v_readlane so the
// gather base is scalar (saddr form). Squared distances computed 16-at-a-time
// as diag(Delta * Delta^T) with v_wmma_f32_16x16x32_f16 from an LDS f16 tile.
__global__ __launch_bounds__(32)
void triplet_tile_kernel(const float* __restrict__ out1,
                         const float* __restrict__ out2,
                         const int*   __restrict__ xy1,
                         const int*   __restrict__ xy2,
                         const int*   __restrict__ nm2,
                         float*       __restrict__ partial)
{
    __shared__ _Float16 dlt[PTS][C_];          // delta tile (f16) for WMMA, 2KB

    const int lane = threadIdx.x;              // 0..31 (wave32)
    const int blk  = blockIdx.x;               // 0..NBLK-1
    const int b    = blk / TILES;
    const int t0   = (blk % TILES) * PTS;

    const size_t imgBase = (size_t)b * C_ * WH_;
    const float* img1 = out1 + imgBase;
    const float* img2 = out2 + imgBase;

    const int    c2     = 2 * lane;            // this lane's channel pair
    const size_t choff0 = (size_t)c2 * WH_;
    const size_t choff1 = choff0 + WH_;

    // ---- gather f1 (2 channels x 16 points) into registers ----
    int xv = 0, yv = 0;
    if (lane < PTS) {
        const int2 c = ((const int2*)xy1)[(size_t)b * N_ + t0 + lane];
        xv = c.x; yv = c.y;
    }
    float f1a[PTS], f1b[PTS];
    #pragma unroll
    for (int i = 0; i < PTS; ++i) {
        const int xi = __builtin_amdgcn_readlane(xv, i);   // SGPR broadcast
        const int yi = __builtin_amdgcn_readlane(yv, i);
        const float* p1 = img1 + (size_t)xi * W_ + yi;     // scalar base
        f1a[i] = p1[choff0];
        f1b[i] = p1[choff1];
    }

    // Build f16 delta tile in LDS vs. gathered out2 features.
    auto build_delta = [&](int x2, int y2) {
        #pragma unroll
        for (int i = 0; i < PTS; ++i) {
            const int xi = __builtin_amdgcn_readlane(x2, i);
            const int yi = __builtin_amdgcn_readlane(y2, i);
            const float* p2 = img2 + (size_t)xi * W_ + yi;
            h2 d;
            d.x = (_Float16)(f1a[i] - p2[choff0]);
            d.y = (_Float16)(f1b[i] - p2[choff1]);
            *reinterpret_cast<h2*>(&dlt[i][c2]) = d;       // ds_store_b32
        }
    };

    // d2 for point (lane) in lanes 0..15: diag(Delta * Delta^T) via 2 WMMAs.
    auto gram_diag_d2 = [&]() -> float {
        v8f acc = {};
        const int rA     = lane & 15;
        const int aShift = (lane < 16) ? 0 : 8;    // A lane layout (ISA 7.12.2)
        const int bShift = (lane < 16) ? 0 : 8;    // B lane layout
        #pragma unroll
        for (int kc = 0; kc < 2; ++kc) {
            v16h a, bm;
            #pragma unroll
            for (int j = 0; j < 16; ++j) {
                const int ka = kc * 32 + j + aShift + ((j >= 8) ? 8 : 0);
                a[j]  = dlt[rA][ka];               // A[row rA][ka]
                const int kb = kc * 32 + j + bShift + ((j >= 8) ? 8 : 0);
                bm[j] = dlt[rA][kb];               // B[k][n] = Delta[n][k]
            }
            acc = __builtin_amdgcn_wmma_f32_16x16x32_f16(
                      false, a, false, bm, (short)0, acc, false, false);
        }
        // Diagonal r: r<8 at (vgpr r, lane r); r>=8 at (vgpr r-8, lane r+16).
        float v = acc[0];
        #pragma unroll
        for (int d = 1; d < 8; ++d) v = ((lane & 7) == d) ? acc[d] : v;
        // Route diag to lanes 0..15 (one ds_bpermute), no LDS round trip.
        const int src = (lane < 8) ? lane : lane + 16;
        return __shfl(v, src, 32);
    };

    // ---- positive distances ----
    int x2 = 0, y2 = 0;
    if (lane < PTS) {
        const int2 c = ((const int2*)xy2)[(size_t)b * N_ + t0 + lane];
        x2 = c.x; y2 = c.y;
    }
    build_delta(x2, y2);
    __syncthreads();
    const float posDist = __builtin_amdgcn_sqrtf(gram_diag_d2() + EPS_);
    __syncthreads();

    // ---- 8 negative distance sets ----
    float negSum = 0.0f;
    for (int mm = 0; mm < M_; ++mm) {
        int xn = 0, yn = 0;
        if (lane < PTS) {
            const size_t o = ((size_t)b * M_ + mm) * N_ + t0 + lane;
            const int2 c = ((const int2*)nm2)[o];
            xn = c.x; yn = c.y;
        }
        build_delta(xn, yn);
        __syncthreads();
        negSum += __builtin_amdgcn_sqrtf(gram_diag_d2() + EPS_);
        __syncthreads();
    }

    // ---- per-point hinge, wave butterfly sum, one store per tile ----
    const float hv = posDist - negSum * (1.0f / M_) + MARGIN_;
    float contrib = (lane < PTS) ? (hv > 0.0f ? hv : 0.0f) : 0.0f;
    #pragma unroll
    for (int off = 16; off >= 1; off >>= 1)
        contrib += __shfl_xor(contrib, off, 32);
    if (lane == 0) partial[blk] = contrib;
}

// Deterministic single-block reduction of the 1000 tile partials.
__global__ __launch_bounds__(256)
void triplet_reduce_kernel(const float* __restrict__ partial,
                           float* __restrict__ out, int nparts)
{
    __shared__ float buf[256];
    float s = 0.0f;
    for (int i = threadIdx.x; i < nparts; i += 256) s += partial[i];
    buf[threadIdx.x] = s;
    __syncthreads();
    for (int off = 128; off > 0; off >>= 1) {
        if ((int)threadIdx.x < off) buf[threadIdx.x] += buf[threadIdx.x + off];
        __syncthreads();
    }
    if (threadIdx.x == 0) out[0] = buf[0] / (float)(B_ * N_);
}

extern "C" void kernel_launch(void* const* d_in, const int* in_sizes, int n_in,
                              void* d_out, int out_size, void* d_ws, size_t ws_size,
                              hipStream_t stream)
{
    const float* out1 = (const float*)d_in[0];
    const float* out2 = (const float*)d_in[1];
    const int*   xy1  = (const int*)d_in[2];
    const int*   xy2  = (const int*)d_in[3];
    const int*   nm2  = (const int*)d_in[4];
    float*       out  = (float*)d_out;
    float*       part = (float*)d_ws;   // NBLK floats of scratch

    triplet_tile_kernel<<<NBLK, 32, 0, stream>>>(out1, out2, xy1, xy2, nm2, part);
    triplet_reduce_kernel<<<1, 256, 0, stream>>>(part, out, NBLK);
}
